// LATTICEModel_87935160418510
// MI455X (gfx1250) — matrix operand order repaired and checked by reading, not compiled, not run on hardware
//
#include <hip/hip_runtime.h>
#include <math.h>
#include <stdint.h>

#define NU 16384
#define NI 8192
#define NTOT (NU + NI)
#define DM 64
#define TWO_E 1048576   // 2*E directed edges
#define TOPK 10

typedef __bf16 bf16;
typedef __attribute__((ext_vector_type(16))) __bf16 v16bf;
typedef __attribute__((ext_vector_type(8)))  float  v8f;

__device__ __forceinline__ void softmax2(const float* __restrict__ iw, float& w0, float& w1) {
  float a = iw[0], b = iw[1];
  float m = fmaxf(a, b);
  float e0 = __expf(a - m), e1 = __expf(b - m);
  float s = e0 + e1;
  w0 = e0 / s; w1 = e1 / s;
}

// ---------------------------------------------------------------------------
// K1: proj = feat @ W + b, row-l2-normalize, store bf16.  16 rows / block, 64 thr.
// ---------------------------------------------------------------------------
__global__ void proj_norm_kernel(const float* __restrict__ feat, const float* __restrict__ W,
                                 const float* __restrict__ b, bf16* __restrict__ xn, int Din) {
  const int j  = threadIdx.x;          // 0..63 = output column
  const int r0 = blockIdx.x * 16;
  __shared__ float sf[16][128];
  __shared__ float s2[16][64];
  __shared__ float nrm[16];

  float acc[16];
  const float bj = b[j];
#pragma unroll
  for (int rr = 0; rr < 16; ++rr) acc[rr] = bj;

  for (int k0 = 0; k0 < Din; k0 += 128) {
    __syncthreads();
    for (int t = j; t < 16 * 128; t += 64)
      sf[t >> 7][t & 127] = feat[(size_t)(r0 + (t >> 7)) * Din + k0 + (t & 127)];
    __syncthreads();
    for (int kk = 0; kk < 128; ++kk) {
      const float wv = W[(size_t)(k0 + kk) * 64 + j];
#pragma unroll
      for (int rr = 0; rr < 16; ++rr) acc[rr] = fmaf(sf[rr][kk], wv, acc[rr]);
    }
  }

  __syncthreads();
#pragma unroll
  for (int rr = 0; rr < 16; ++rr) s2[rr][j] = acc[rr] * acc[rr];
  __syncthreads();
  if (j < 16) {
    float s = 0.f;
    for (int t = 0; t < 64; ++t) s += s2[j][t];
    nrm[j] = sqrtf(s);
  }
  __syncthreads();
#pragma unroll
  for (int rr = 0; rr < 16; ++rr)
    xn[(size_t)(r0 + rr) * 64 + j] = (bf16)(acc[rr] / nrm[rr]);
}

// ---------------------------------------------------------------------------
// K2: transpose-convert Gi (8192x64 f32) -> giT (64x8192 bf16) for GEMM B-operand
// ---------------------------------------------------------------------------
__global__ void transpose_gi_kernel(const float* __restrict__ Gi, bf16* __restrict__ giT) {
  const int t = blockIdx.x * 256 + threadIdx.x;   // NI*64 threads
  const int r = t >> 6, c = t & 63;
  giT[(size_t)c * NI + r] = (bf16)Gi[t];
}

// ---------------------------------------------------------------------------
// K3: fused  sim = xn@xn^T  +  per-row top-10.  Block=128 (4 waves), each wave
// owns a 16-row strip; column tile (16x64 bf16) is DOUBLE-BUFFERED in LDS and
// filled with the gfx1250 async-copy path (global_load_async_to_lds_b128,
// ASYNCcnt), overlapping the next tile fetch with WMMA + the top-k scan.
// WMMA: v_wmma_f32_16x16x32_bf16, 2 per 16x16 tile (K=64).
// grid = (NI/64, 2 modalities)
// ---------------------------------------------------------------------------
__global__ void sim_topk_kernel(const bf16* __restrict__ xnv, const bf16* __restrict__ xnt,
                                float* __restrict__ tkv, int* __restrict__ tki) {
  const bf16* __restrict__ xn = blockIdx.y ? xnt : xnv;
  const int tid  = threadIdx.x;
  const int wave = tid >> 5;
  const int lane = tid & 31;
  const int m    = lane & 15;          // row within strip (A) / col within tile (B)
  const int hi   = lane >> 4;          // lane half
  const int base = hi * 16;            // A K-offset for this half
  const int rowBase = blockIdx.x * 64 + wave * 16;
  const int row     = rowBase + m;

  __shared__ __align__(32) bf16 colT[2][16 * 64];    // double-buffered column tile
  __shared__ float cbuf[4][16 * 16];                 // per-wave C tile
  __shared__ float mgV[4][16 * 20];                  // per-wave merge buffers
  __shared__ int   mgI[4][16 * 20];

  // Preload A strip (row `row`, K=0..63) into two v16bf, contiguous K order (k = 16*half + s).
  v16bf a0, a1;
  const bf16* rp = xn + (size_t)row * 64;
#pragma unroll
  for (int e = 0; e < 8; ++e) {
    a0[e]     = rp[base + e];
    a0[8 + e] = rp[base + 8 + e];
    a1[e]     = rp[32 + base + e];
    a1[8 + e] = rp[32 + base + 8 + e];
  }

  float vals[TOPK];
  int   idxs[TOPK];
#pragma unroll
  for (int s = 0; s < TOPK; ++s) { vals[s] = -1e30f; idxs[s] = 0; }
  float minv = -1e30f;

  // async-stage tile `ct` into colT[buf]: 128 lanes x 16B = full 2KB tile,
  // written directly into LDS by the async DMA path (no VGPR round trip).
  auto issue_tile = [&](int ct, int buf) {
    const bf16* g = xn + (size_t)ct * 1024 + tid * 8;
    const uint32_t lds = (uint32_t)(uintptr_t)&colT[buf][tid * 8];
    asm volatile("global_load_async_to_lds_b128 %0, %1, off"
                 :: "v"(lds), "v"(g) : "memory");
  };

  issue_tile(0, 0);
  for (int ct = 0; ct < NI / 16; ++ct) {
    const int cur = ct & 1;
    asm volatile("s_wait_asynccnt 0" ::: "memory");   // my async writes landed
    __syncthreads();                                  // everyone's landed
    if (ct + 1 < NI / 16) issue_tile(ct + 1, cur ^ 1);

    v8f c = {0.f, 0.f, 0.f, 0.f, 0.f, 0.f, 0.f, 0.f};
    const v16bf b0 = *(const v16bf*)&colT[cur][m * 64 + hi * 16];
    const v16bf b1 = *(const v16bf*)&colT[cur][m * 64 + 32 + hi * 16];
    c = __builtin_amdgcn_wmma_f32_16x16x32_bf16(false, a0, false, b0, (short)0, c, false, false);
    c = __builtin_amdgcn_wmma_f32_16x16x32_bf16(false, a1, false, b1, (short)0, c, false, false);

#pragma unroll
    for (int v = 0; v < 8; ++v) cbuf[wave][(v + hi * 8) * 16 + m] = c[v];
    __syncthreads();

    // each lane scans 8 columns of its row; cached-min threshold makes the
    // full insertion path rare.
#pragma unroll
    for (int jj = 0; jj < 8; ++jj) {
      const float vv = cbuf[wave][m * 16 + hi * 8 + jj];
      if (vv > minv) {
        int ms = 0; float mv = vals[0];
#pragma unroll
        for (int s = 1; s < TOPK; ++s) if (vals[s] < mv) { mv = vals[s]; ms = s; }
#pragma unroll
        for (int s = 0; s < TOPK; ++s) if (s == ms) { vals[s] = vv; idxs[s] = ct * 16 + hi * 8 + jj; }
        minv = vals[0];
#pragma unroll
        for (int s = 1; s < TOPK; ++s) minv = fminf(minv, vals[s]);
      }
    }
  }

  // merge the two half-lane partial top-10 lists per row
#pragma unroll
  for (int s = 0; s < TOPK; ++s) {
    mgV[wave][m * 20 + hi * 10 + s] = vals[s];
    mgI[wave][m * 20 + hi * 10 + s] = idxs[s];
  }
  __syncthreads();
  if (hi == 0) {
    for (int s = 0; s < TOPK; ++s) {
      float best = -1e30f; int bi = 0;
      for (int t = 0; t < 20; ++t) {
        const float q = mgV[wave][m * 20 + t];
        if (q > best) { best = q; bi = t; }
      }
      tkv[((size_t)blockIdx.y * NI + row) * TOPK + s] = best;
      tki[((size_t)blockIdx.y * NI + row) * TOPK + s] = mgI[wave][m * 20 + bi];
      mgV[wave][m * 20 + bi] = -1e30f;
    }
  }
}

// ---------------------------------------------------------------------------
// K4: degree -> dis = rsqrt(rowsum(learned)) for the normalized Laplacian
// ---------------------------------------------------------------------------
__global__ void lap_dis_kernel(const float* __restrict__ tkv, const float* __restrict__ iw,
                               float* __restrict__ dis) {
  const int r = blockIdx.x * 256 + threadIdx.x;
  if (r >= NI) return;
  float w0, w1; softmax2(iw, w0, w1);
  float d = 0.f;
#pragma unroll
  for (int s = 0; s < TOPK; ++s)
    d += w0 * tkv[(size_t)r * TOPK + s] + w1 * tkv[((size_t)NI + r) * TOPK + s];
  dis[r] = d > 0.f ? rsqrtf(d) : 0.f;
}

// ---------------------------------------------------------------------------
// K5: sparse part of item_emb: 0.3 * Lap(learned) @ Gi   (<=20 nnz per row)
// ---------------------------------------------------------------------------
__global__ void item_learned_kernel(const float* __restrict__ tkv, const int* __restrict__ tki,
                                    const float* __restrict__ dis, const float* __restrict__ Gi,
                                    const float* __restrict__ iw, float* __restrict__ item) {
  const int r = blockIdx.x, j = threadIdx.x;
  float w0, w1; softmax2(iw, w0, w1);
  const float dr = dis[r];
  float acc = 0.f;
#pragma unroll
  for (int mm = 0; mm < 2; ++mm) {
    const float wm = mm ? w1 : w0;
#pragma unroll
    for (int s = 0; s < TOPK; ++s) {
      const size_t o = ((size_t)mm * NI + r) * TOPK + s;
      const int c = tki[o];
      const float coef = 0.3f * wm * tkv[o] * dr * dis[c];
      acc = fmaf(coef, Gi[(size_t)c * 64 + j], acc);
    }
  }
  item[(size_t)r * 64 + j] = acc;
}

// ---------------------------------------------------------------------------
// K6: dense part: item += 0.7 * (w0*Sim_v + w1*Sim_t) @ Gi.
// One wave per 16 output rows; A combined+converted to bf16 on the fly.
// This kernel is the HBM roofline bottleneck (2 x 256MB Sim reads ~ 22us at
// 23.3 TB/s) -> prefetch the row streams 1KB ahead (global_prefetch_b8).
// ---------------------------------------------------------------------------
__global__ void dense_item_kernel(const float* __restrict__ Sv, const float* __restrict__ St,
                                  const bf16* __restrict__ giT, const float* __restrict__ iw,
                                  float* __restrict__ item) {
  const int lane = threadIdx.x;       // 32 threads = 1 wave
  const int m    = lane & 15;
  const int hi   = lane >> 4;
  const int base = hi * 16;
  const int rowBase = blockIdx.x * 16;
  float w0, w1; softmax2(iw, w0, w1);

  const float* __restrict__ rv = Sv + (size_t)(rowBase + m) * NI;
  const float* __restrict__ rt = St + (size_t)(rowBase + m) * NI;

  v8f c0 = {0,0,0,0,0,0,0,0}, c1 = c0, c2 = c0, c3 = c0;

  for (int kb = 0; kb < NI; kb += 32) {
    if (kb + 256 < NI) {
      __builtin_prefetch(rv + kb + 256, 0, 1);
      __builtin_prefetch(rt + kb + 256, 0, 1);
    }
    v16bf a;
#pragma unroll
    for (int e = 0; e < 8; ++e) {
      a[e]     = (bf16)fmaf(w0, rv[kb + base + e],      w1 * rt[kb + base + e]);
      a[8 + e] = (bf16)fmaf(w0, rv[kb + base + 8 + e], w1 * rt[kb + base + 8 + e]);
    }
    const v16bf b0 = *(const v16bf*)&giT[(size_t)(0 * 16 + m) * NI + kb + hi * 16];
    const v16bf b1 = *(const v16bf*)&giT[(size_t)(1 * 16 + m) * NI + kb + hi * 16];
    const v16bf b2 = *(const v16bf*)&giT[(size_t)(2 * 16 + m) * NI + kb + hi * 16];
    const v16bf b3 = *(const v16bf*)&giT[(size_t)(3 * 16 + m) * NI + kb + hi * 16];
    c0 = __builtin_amdgcn_wmma_f32_16x16x32_bf16(false, a, false, b0, (short)0, c0, false, false);
    c1 = __builtin_amdgcn_wmma_f32_16x16x32_bf16(false, a, false, b1, (short)0, c1, false, false);
    c2 = __builtin_amdgcn_wmma_f32_16x16x32_bf16(false, a, false, b2, (short)0, c2, false, false);
    c3 = __builtin_amdgcn_wmma_f32_16x16x32_bf16(false, a, false, b3, (short)0, c3, false, false);
  }

#pragma unroll
  for (int v = 0; v < 8; ++v) {
    const size_t ro = (size_t)(rowBase + v + hi * 8) * 64 + m;
    item[ro + 0]  += 0.7f * c0[v];
    item[ro + 16] += 0.7f * c1[v];
    item[ro + 32] += 0.7f * c2[v];
    item[ro + 48] += 0.7f * c3[v];
  }
}

// ---------------------------------------------------------------------------
// K7: row-wise l2 normalize (x / max(||x||, 1e-12)); grid = rows, block = 64
// ---------------------------------------------------------------------------
__global__ void l2norm_rows_kernel(const float* __restrict__ in, float* __restrict__ out) {
  const int r = blockIdx.x, j = threadIdx.x;
  const float v = in[(size_t)r * 64 + j];
  __shared__ float s[64];
  s[j] = v * v;
  __syncthreads();
  for (int off = 32; off > 0; off >>= 1) {
    if (j < off) s[j] += s[j + off];
    __syncthreads();
  }
  const float n = sqrtf(s[0]);
  out[(size_t)r * 64 + j] = v / fmaxf(n, 1e-12f);
}

__global__ void zero_kernel(float* __restrict__ p, int n) {
  const int t = blockIdx.x * 256 + threadIdx.x;
  if (t < n) p[t] = 0.f;
}

// ---------------------------------------------------------------------------
// K8: edge scatter  agg[dst] += emb[src]; 16 threads x 4 dims per edge.
// agg (6MB) is L2-resident on the 192MB L2, so the f32 atomics stay on-chip.
// ---------------------------------------------------------------------------
__global__ void scatter_kernel(const int* __restrict__ ei, const float* __restrict__ embU,
                               const float* __restrict__ embI, float* __restrict__ agg) {
  const size_t t = (size_t)blockIdx.x * 256 + threadIdx.x;   // TWO_E*16 threads
  const size_t e = t >> 4;
  const int q = (int)(t & 15) * 4;
  const int src = ei[e];
  const int dst = ei[TWO_E + e];
  const float* sp = (src < NU) ? (embU + (size_t)src * 64) : (embI + (size_t)(src - NU) * 64);
  const float4 v = *(const float4*)(sp + q);
  float* dp = agg + (size_t)dst * 64 + q;
  atomicAdd(dp + 0, v.x);
  atomicAdd(dp + 1, v.y);
  atomicAdd(dp + 2, v.z);
  atomicAdd(dp + 3, v.w);
}

// ---------------------------------------------------------------------------
// K9: final mean + item add. d_out = [gu | gi + l2norm(item)] (contiguous).
// ---------------------------------------------------------------------------
__global__ void final_kernel(const float* __restrict__ Gu, const float* __restrict__ Gi,
                             const float* __restrict__ e1, const float* __restrict__ e2,
                             const float* __restrict__ itemn, float* __restrict__ out) {
  const int r = blockIdx.x, j = threadIdx.x;
  const size_t o = (size_t)r * 64 + j;
  const float b = (r < NU) ? Gu[o] : Gi[(size_t)(r - NU) * 64 + j];
  float v = (b + e1[o] + e2[o]) * (1.0f / 3.0f);
  if (r >= NU) v += itemn[(size_t)(r - NU) * 64 + j];
  out[o] = v;
}

// ---------------------------------------------------------------------------
extern "C" void kernel_launch(void* const* d_in, const int* in_sizes, int n_in,
                              void* d_out, int out_size, void* d_ws, size_t ws_size,
                              hipStream_t stream) {
  const float* Gu     = (const float*)d_in[0];
  const float* Gi     = (const float*)d_in[1];
  const float* feat_v = (const float*)d_in[2];
  const float* feat_t = (const float*)d_in[3];
  const float* W_v    = (const float*)d_in[4];
  const float* b_v    = (const float*)d_in[5];
  const float* W_t    = (const float*)d_in[6];
  const float* b_t    = (const float*)d_in[7];
  const float* iw     = (const float*)d_in[8];
  const float* Sim_v  = (const float*)d_in[9];
  const float* Sim_t  = (const float*)d_in[10];
  const int*   ei     = (const int*)d_in[11];
  float* out = (float*)d_out;

  char* p = (char*)d_ws;
  auto alloc = [&](size_t bytes) -> void* {
    void* q = (void*)p;
    p += (bytes + 255) & ~(size_t)255;
    return q;
  };
  bf16*  xnv  = (bf16*)alloc((size_t)NI * 64 * sizeof(bf16));
  bf16*  xnt  = (bf16*)alloc((size_t)NI * 64 * sizeof(bf16));
  bf16*  giT  = (bf16*)alloc((size_t)64 * NI * sizeof(bf16));
  float* tkv  = (float*)alloc((size_t)2 * NI * TOPK * sizeof(float));
  int*   tki  = (int*)  alloc((size_t)2 * NI * TOPK * sizeof(int));
  float* dis  = (float*)alloc((size_t)NI * sizeof(float));
  float* item = (float*)alloc((size_t)NI * 64 * sizeof(float));
  float* agg  = (float*)alloc((size_t)NTOT * 64 * sizeof(float));
  float* emb1 = (float*)alloc((size_t)NTOT * 64 * sizeof(float));
  float* emb2 = (float*)alloc((size_t)NTOT * 64 * sizeof(float));

  // projections + row-normalize (bf16 out)
  proj_norm_kernel<<<NI / 16, 64, 0, stream>>>(feat_v, W_v, b_v, xnv, 4096);
  proj_norm_kernel<<<NI / 16, 64, 0, stream>>>(feat_t, W_t, b_t, xnt, 384);
  transpose_gi_kernel<<<(NI * 64) / 256, 256, 0, stream>>>(Gi, giT);

  // fused similarity GEMM + top-k, both modalities in one launch
  sim_topk_kernel<<<dim3(NI / 64, 2), 128, 0, stream>>>(xnv, xnt, tkv, tki);

  // learned-graph Laplacian + item propagation (sparse + dense WMMA parts)
  lap_dis_kernel<<<(NI + 255) / 256, 256, 0, stream>>>(tkv, iw, dis);
  item_learned_kernel<<<NI, 64, 0, stream>>>(tkv, tki, dis, Gi, iw, item);
  dense_item_kernel<<<NI / 16, 32, 0, stream>>>(Sim_v, Sim_t, giT, iw, item);
  l2norm_rows_kernel<<<NI, 64, 0, stream>>>(item, item);

  // bipartite UI propagation: 2 layers of scatter + normalize
  zero_kernel<<<(NTOT * 64) / 256, 256, 0, stream>>>(agg, NTOT * 64);
  scatter_kernel<<<(TWO_E * 16) / 256, 256, 0, stream>>>(ei, Gu, Gi, agg);
  l2norm_rows_kernel<<<NTOT, 64, 0, stream>>>(agg, emb1);

  zero_kernel<<<(NTOT * 64) / 256, 256, 0, stream>>>(agg, NTOT * 64);
  scatter_kernel<<<(TWO_E * 16) / 256, 256, 0, stream>>>(ei, emb1, emb1 + (size_t)NU * 64, agg);
  l2norm_rows_kernel<<<NTOT, 64, 0, stream>>>(agg, emb2);

  final_kernel<<<NTOT, 64, 0, stream>>>(Gu, Gi, emb1, emb2, item, out);
}
